// EIRNN_33337536151734
// MI455X (gfx1250) — matrix operand order repaired and checked by reading, not compiled, not run on hardware
//
#include <hip/hip_runtime.h>

// ---------------------------------------------------------------------------
// EI-RNN scan for gfx1250 (MI455X) — persistent-kernel version, round 3.
//   per step: total = x_t @ W_in^T + out_{t-1} @ W_eff^T + (b_in + b_rec)
//             state = 0.8*state + 0.2*total + SIG*noise ; out = relu(state)
//
// Design:
//  * ONE persistent kernel runs all 1024 timesteps (no per-step launch cost).
//  * 64 blocks x 128 threads (4 wave32s); block owns a fixed 64(M)x32(N) tile.
//  * fp32 state kept in REGISTERS across the whole scan (saves ~1 GB traffic).
//  * Recurrent weights (16 B-fragments = 128 VGPRs/wave) hoisted into
//    REGISTERS for the whole scan -> recurrent WMMAs need no operand loads.
//  * Input-weight slice (8 KB) cached in LDS; B-fragments via ds_load_b128.
//  * x fp32->bf16 via v_cvt_pk_bf16_f32 (hardware RTE) when available.
//  * bf16 operands + fp32 accumulation: v_wmma_f32_16x16x32_bf16.
//  * Steps separated by a grid-wide monotonic barrier (atomicAdd + s_sleep
//    spin, agent-scope release/acquire fences) over a bf16 out ping-pong.
// ---------------------------------------------------------------------------

#define SEQL   1024
#define BATCH  512
#define INSZ   128
#define HID    256
#define ALPHA_ 0.2f
#define OMA_   0.8f
#define SIG_   0.03162277660168379f   // sqrt(2*0.2)*0.05
#define NBLK   64                     // 8 row-blocks x 8 col-blocks

typedef __attribute__((ext_vector_type(8)))  float          v8f;
typedef __attribute__((ext_vector_type(16))) __bf16         v16bf;
typedef __attribute__((ext_vector_type(8)))  __bf16         v8bf;
typedef __attribute__((ext_vector_type(2)))  __bf16         v2bf;
typedef __attribute__((ext_vector_type(8)))  unsigned short v8u;
typedef unsigned short ushort_t;

static __device__ __forceinline__ unsigned short f2bf_rte(float f) {
    unsigned int u = __float_as_uint(f);
    u = (u + 0x7FFFu + ((u >> 16) & 1u)) >> 16;   // round-to-nearest-even
    return (unsigned short)u;
}

static __device__ __forceinline__ v16bf make_frag(v8bf lo, v8bf hi) {
    union { v16bf v; v8bf h[2]; } u;
    u.h[0] = lo; u.h[1] = hi;
    return u.v;
}

// 8 x fp32 -> 8 x bf16 (RTE). Prefer the packed hardware convert.
#if __has_builtin(__builtin_amdgcn_cvt_pk_bf16_f32)
static __device__ __forceinline__ v8bf cvt8f(float4 a, float4 b) {
    union { v8bf v; v2bf p[4]; } u;
    u.p[0] = __builtin_bit_cast(v2bf, __builtin_amdgcn_cvt_pk_bf16_f32(a.x, a.y));
    u.p[1] = __builtin_bit_cast(v2bf, __builtin_amdgcn_cvt_pk_bf16_f32(a.z, a.w));
    u.p[2] = __builtin_bit_cast(v2bf, __builtin_amdgcn_cvt_pk_bf16_f32(b.x, b.y));
    u.p[3] = __builtin_bit_cast(v2bf, __builtin_amdgcn_cvt_pk_bf16_f32(b.z, b.w));
    return u.v;
}
#else
static __device__ __forceinline__ v8bf cvt8f(float4 a, float4 b) {
    v8bf r;
    r[0] = (__bf16)a.x; r[1] = (__bf16)a.y; r[2] = (__bf16)a.z; r[3] = (__bf16)a.w;
    r[4] = (__bf16)b.x; r[5] = (__bf16)b.y; r[6] = (__bf16)b.z; r[7] = (__bf16)b.w;
    return r;
}
#endif

// Grid-wide barrier: monotonic counter, target = (step+1)*NBLK. Release fence
// before arrive (flush this WGP's writes), acquire fence after (invalidate L0
// so next step sees the other WGPs' bf16 activations).
static __device__ __forceinline__ void grid_barrier(unsigned* bar, unsigned target) {
    __syncthreads();
    if (threadIdx.x == 0) {
        __builtin_amdgcn_fence(__ATOMIC_RELEASE, "agent");
        __hip_atomic_fetch_add(bar, 1u, __ATOMIC_RELAXED, __HIP_MEMORY_SCOPE_AGENT);
        while (__hip_atomic_load(bar, __ATOMIC_RELAXED, __HIP_MEMORY_SCOPE_AGENT) < target) {
            __builtin_amdgcn_s_sleep(1);
        }
        __builtin_amdgcn_fence(__ATOMIC_ACQUIRE, "agent");
    }
    __syncthreads();
}

// ---------------------------------------------------------------------------
// One-time setup: w_eff = |w_rec|*mask -> bf16 ; w_in -> bf16 ; bias = b_in+b_rec;
// zero the t=0 recurrent input (out_{-1}==0); reset the grid barrier counter.
// ---------------------------------------------------------------------------
__global__ void eirnn_setup(const float* __restrict__ w_in,
                            const float* __restrict__ b_in,
                            const float* __restrict__ w_rec,
                            const float* __restrict__ b_rec,
                            const float* __restrict__ mask,
                            ushort_t* __restrict__ w_in_bf,
                            ushort_t* __restrict__ w_eff_bf,
                            float* __restrict__ bias,
                            ushort_t* __restrict__ outbuf0,
                            unsigned* __restrict__ bar) {
    int i = blockIdx.x * blockDim.x + threadIdx.x;        // 0 .. 262143
    if (i == 0)          *bar = 0u;                       // deterministic per call
    if (i < HID * HID)   w_eff_bf[i] = f2bf_rte(fabsf(w_rec[i]) * mask[i]);
    if (i < HID * INSZ)  w_in_bf[i]  = f2bf_rte(w_in[i]);
    if (i < HID)         bias[i]     = b_in[i] + b_rec[i];
    if (i < BATCH * HID) outbuf0[i]  = 0;
}

// ---------------------------------------------------------------------------
// Persistent scan kernel: grid = 64 blocks, block = 128 threads (4 wave32s).
// Each wave computes a 16(M) x 32(N) strip: 2 f32 accumulators per step,
// combined K = 128 (input) + 256 (recurrent), 24 WMMAs per wave per step.
// ---------------------------------------------------------------------------
__global__ __launch_bounds__(128) void eirnn_scan(
        const float*    __restrict__ x,         // [SEQ][BATCH][INSZ] fp32
        const float*    __restrict__ noise,     // [SEQ][BATCH][HID]  fp32
        const ushort_t* __restrict__ w_in_bf,   // [HID][INSZ]  bf16
        const ushort_t* __restrict__ w_eff_bf,  // [HID][HID]   bf16
        const float*    __restrict__ bias,      // [HID]
        ushort_t*       __restrict__ buf0,      // [BATCH][HID] bf16 ping
        ushort_t*       __restrict__ buf1,      // [BATCH][HID] bf16 pong
        unsigned*       __restrict__ bar,       // grid barrier counter
        float*          __restrict__ outs,      // [SEQ][BATCH][HID] fp32
        float*          __restrict__ state_f,   // [BATCH][HID] fp32
        float*          __restrict__ out_f)     // [BATCH][HID] fp32
{
    const int lane    = threadIdx.x & 31;
    const int wave    = threadIdx.x >> 5;
    const int mblock  = (blockIdx.x & 7) * 64;         // batch-row block
    const int n0      = (blockIdx.x >> 3) * 32;        // hidden-col block
    const int m0      = mblock + wave * 16;
    const int halfsel = lane >> 4;                     // 0: lanes 0-15, 1: 16-31
    const int mrow    = lane & 15;

    const int arow  = m0 + mrow;                       // A fragment row (batch)
    const int asel  = halfsel * 8;                     // A K-subselect per half
    const int bsel  = halfsel * 8;                     // B K-subselect per half
    const int bcol0 = n0 + mrow;                       // global cols of accums
    const int bcol1 = n0 + 16 + mrow;
    const int lcol0 = mrow;                            // LDS-local columns
    const int lcol1 = 16 + mrow;

    // --- cache this block's input-weight slice in LDS (once, reused 1024x) ---
    __shared__ ushort_t smem_win[32 * INSZ];           // 8 KB
    {
        const ushort_t* gin = w_in_bf + (size_t)n0 * INSZ;   // rows n0..n0+31
        for (int j = threadIdx.x * 8; j < 32 * INSZ; j += 128 * 8)
            *(v8u*)(smem_win + j) = *(const v8u*)(gin + j);
    }
    __syncthreads();

    // --- recurrent weights: 16 B-fragments held in REGISTERS for the scan ---
    v16bf wr0[8], wr1[8];
#pragma unroll
    for (int c = 0; c < 8; ++c) {
        const ushort_t* b0p = w_eff_bf + (size_t)bcol0 * HID + c * 32 + bsel;
        const ushort_t* b1p = w_eff_bf + (size_t)bcol1 * HID + c * 32 + bsel;
        wr0[c] = make_frag(*(const v8bf*)b0p, *(const v8bf*)(b0p + 16));
        wr1[c] = make_frag(*(const v8bf*)b1p, *(const v8bf*)(b1p + 16));
    }

    const float bsum0 = bias[bcol0];
    const float bsum1 = bias[bcol1];

    // fp32 state lives in registers for the entire scan
    float st0[8], st1[8];
#pragma unroll
    for (int r = 0; r < 8; ++r) { st0[r] = 0.0f; st1[r] = 0.0f; }

#pragma unroll 1
    for (int t = 0; t < SEQL; ++t) {
        const ushort_t* prev = (t & 1) ? buf1 : buf0;
        ushort_t*       next = (t & 1) ? buf0 : buf1;
        const float* x_t     = x     + (size_t)t * BATCH * INSZ;
        const float* noise_t = noise + (size_t)t * BATCH * HID;
        float*       outs_t  = outs  + (size_t)t * BATCH * HID;

        // early prefetch of this step's streaming inputs (global_prefetch_b8)
        {
            int r = threadIdx.x & 63;
            if (threadIdx.x < 64)
                __builtin_prefetch(noise_t + (size_t)(mblock + r) * HID + n0, 0, 0);
            else
                __builtin_prefetch(x_t + (size_t)(mblock + r) * INSZ, 0, 0);
        }

        v8f acc0 = {};   // cols n0 .. n0+15
        v8f acc1 = {};   // cols n0+16 .. n0+31

        // ---- input contribution: K = 128 (x fp32 -> bf16 via cvt_pk) ----
        const float* xrow = x_t + (size_t)arow * INSZ;
#pragma unroll
        for (int kc = 0; kc < INSZ; kc += 32) {
            const float4* ap = (const float4*)(xrow + kc + asel);
            v16bf afrag = make_frag(cvt8f(ap[0], ap[1]),    // K = kc+asel   ..+7
                                    cvt8f(ap[4], ap[5]));   // K = kc+asel+16..+23
            const ushort_t* b0p = smem_win + lcol0 * INSZ + kc + bsel;
            const ushort_t* b1p = smem_win + lcol1 * INSZ + kc + bsel;
            v16bf bf0 = make_frag(*(const v8bf*)b0p, *(const v8bf*)(b0p + 16));
            v16bf bf1 = make_frag(*(const v8bf*)b1p, *(const v8bf*)(b1p + 16));
            acc0 = __builtin_amdgcn_wmma_f32_16x16x32_bf16(false, afrag, false, bf0,
                                                           (short)0, acc0, false, false);
            acc1 = __builtin_amdgcn_wmma_f32_16x16x32_bf16(false, afrag, false, bf1,
                                                           (short)0, acc1, false, false);
        }

        // ---- recurrent contribution: K = 256; B straight from registers ----
        const ushort_t* orow = prev + (size_t)arow * HID;
#pragma unroll
        for (int c = 0; c < 8; ++c) {
            const ushort_t* ap = orow + c * 32 + asel;
            v16bf afrag = make_frag(*(const v8bf*)ap, *(const v8bf*)(ap + 16));
            acc0 = __builtin_amdgcn_wmma_f32_16x16x32_bf16(false, afrag, false, wr0[c],
                                                           (short)0, acc0, false, false);
            acc1 = __builtin_amdgcn_wmma_f32_16x16x32_bf16(false, afrag, false, wr1[c],
                                                           (short)0, acc1, false, false);
        }

        // ---- fused epilogue: bias, leaky update, noise, relu, stores ----
        __bf16* nextb = (__bf16*)next;
#pragma unroll
        for (int r = 0; r < 8; ++r) {
            const int row   = m0 + r + halfsel * 8;    // C/D layout: M = r / r+8
            const size_t i0 = (size_t)row * HID + bcol0;
            const size_t i1 = (size_t)row * HID + bcol1;
            float s0 = st0[r] * OMA_ + (acc0[r] + bsum0) * ALPHA_ + SIG_ * noise_t[i0];
            float s1 = st1[r] * OMA_ + (acc1[r] + bsum1) * ALPHA_ + SIG_ * noise_t[i1];
            st0[r] = s0;
            st1[r] = s1;
            float o0 = fmaxf(s0, 0.0f);
            float o1 = fmaxf(s1, 0.0f);
            outs_t[i0] = o0;
            outs_t[i1] = o1;
            nextb[i0]  = (__bf16)o0;
            nextb[i1]  = (__bf16)o1;
        }

        grid_barrier(bar, (unsigned)(t + 1) * NBLK);
    }

    // ---- final state / final output straight from registers ----
#pragma unroll
    for (int r = 0; r < 8; ++r) {
        const int row   = m0 + r + halfsel * 8;
        const size_t i0 = (size_t)row * HID + bcol0;
        const size_t i1 = (size_t)row * HID + bcol1;
        state_f[i0] = st0[r];
        state_f[i1] = st1[r];
        out_f[i0]   = fmaxf(st0[r], 0.0f);
        out_f[i1]   = fmaxf(st1[r], 0.0f);
    }
}

// ---------------------------------------------------------------------------
// Host side. ws layout (bytes, offsets 1KB-aligned):
//   [0       ..  65536) w_in  bf16 [HID][INSZ]
//   [65536   .. 196608) w_eff bf16 [HID][HID]
//   [196608  .. 197632) bias  fp32 [HID]
//   [197632  .. 459776) out ping bf16 [BATCH][HID]
//   [459776  .. 721920) out pong bf16 [BATCH][HID]
//   [721920  .. 722944) grid barrier counter
// ---------------------------------------------------------------------------
extern "C" void kernel_launch(void* const* d_in, const int* in_sizes, int n_in,
                              void* d_out, int out_size, void* d_ws, size_t ws_size,
                              hipStream_t stream) {
    const float* x     = (const float*)d_in[0];
    const float* w_in  = (const float*)d_in[1];
    const float* b_in  = (const float*)d_in[2];
    const float* w_rec = (const float*)d_in[3];
    const float* b_rec = (const float*)d_in[4];
    const float* mask  = (const float*)d_in[5];
    const float* noise = (const float*)d_in[6];
    float* out = (float*)d_out;

    char* ws = (char*)d_ws;
    ushort_t* w_in_bf  = (ushort_t*)(ws);
    ushort_t* w_eff_bf = (ushort_t*)(ws + 65536);
    float*    bias     = (float*)   (ws + 196608);
    ushort_t* buf0     = (ushort_t*)(ws + 197632);
    ushort_t* buf1     = (ushort_t*)(ws + 459776);
    unsigned* bar      = (unsigned*)(ws + 721920);

    eirnn_setup<<<1024, 256, 0, stream>>>(w_in, b_in, w_rec, b_rec, mask,
                                          w_in_bf, w_eff_bf, bias, buf0, bar);

    const size_t outs_elems = (size_t)SEQL * BATCH * HID;
    eirnn_scan<<<NBLK, 128, 0, stream>>>(
        x, noise, w_in_bf, w_eff_bf, bias, buf0, buf1, bar,
        out,                                   // outs  (SEQ,B,H)
        out + outs_elems,                      // state_f (B,H)
        out + outs_elems + (size_t)BATCH * HID /* out_f (B,H) */);
}
